// FleetEncoder_40183714021849
// MI455X (gfx1250) — matrix operand order repaired and checked by reading, not compiled, not run on hardware
//
#include <hip/hip_runtime.h>
#include <hip/hip_bf16.h>
#include <math.h>

typedef _Float16 half_t;
typedef __attribute__((ext_vector_type(16))) _Float16 v16h;
typedef __attribute__((ext_vector_type(8)))  _Float16 v8h;
typedef __attribute__((ext_vector_type(8)))  float    v8f;

// ---- problem constants (from reference) ----
constexpr int N_  = 64;
constexpr int L_  = 128;
constexpr int DV_ = 16;
constexpr int EF_ = 4;
constexpr int D_  = 256;
constexpr int H_  = 8;
constexpr int DK_ = 32;
constexpr int FF_ = 1024;
constexpr int LC_ = 3;
constexpr int KNN_ = 6;
constexpr int MS_ = 16;
constexpr int R_  = N_ * L_;          // 8192 rows
constexpr float INV_SQRT_DK = 0.17677669529663687f;  // 1/sqrt(32)

// =====================================================================
// pack dist = fleet_edges[..., 0] into contiguous (N, L, L) f32
// =====================================================================
__global__ void fe_pack_dist(const float* __restrict__ fe, float* __restrict__ dist) {
    const int i = blockIdx.x * blockDim.x + threadIdx.x;
    if (i < N_ * L_ * L_) dist[i] = fe[(size_t)i * EF_];
}

// =====================================================================
// f32 -> f16 convert with transpose: src (batch, Kd, Nc) -> dst (batch, Nc, Kd)
// so WMMA B fragments become contiguous v16h loads (N-major weights).
// =====================================================================
__global__ void fe_convert_f16_t(const float* __restrict__ src, half_t* __restrict__ dst,
                                 int Kd, int Nc) {
    const int i = blockIdx.x * blockDim.x + threadIdx.x;
    if (i >= Kd * Nc) return;
    const int k  = i / Nc;
    const int nc = i - k * Nc;
    const size_t base = (size_t)blockIdx.y * Kd * Nc;
    dst[base + (size_t)nc * Kd + k] = (half_t)src[base + i];
}

// =====================================================================
// input projection: h = vehicles @ Wproj + bproj   (K=16, scalar f32)
// =====================================================================
__global__ void fe_proj(const float* __restrict__ veh, const float* __restrict__ Wp,
                        const float* __restrict__ bp, float* __restrict__ h32,
                        half_t* __restrict__ h16) {
    int idx = blockIdx.x * blockDim.x + threadIdx.x;
    if (idx >= R_ * D_) return;
    int row = idx >> 8;          // / D_
    int col = idx & (D_ - 1);
    const float* v = veh + row * DV_;
    float s = bp[col];
#pragma unroll
    for (int j = 0; j < DV_; ++j) s += v[j] * Wp[j * D_ + col];
    h32[idx] = s;
    h16[idx] = (half_t)s;
}

// =====================================================================
// kNN: 6 smallest distances per row (ties -> lower index), emit indices.
// Softmax with -1e9 masking is exactly softmax over these 6 columns
// (expf(-1e9 - mx) underflows to 0 in f32), so downstream is sparse.
// =====================================================================
__global__ void fe_knn(const float* __restrict__ dist, int* __restrict__ kidx) {
    int idx = blockIdx.x * blockDim.x + threadIdx.x;   // n*L + r
    if (idx >= N_ * L_) return;
    const float* row = dist + (size_t)idx * L_;
    float bv[KNN_]; int bi[KNN_];
#pragma unroll
    for (int j = 0; j < KNN_; ++j) { bv[j] = 3.4e38f; bi[j] = 0; }
    for (int c = 0; c < L_; ++c) {
        float d = row[c];
        if (d < bv[KNN_ - 1]) {              // strict <: equal values keep earlier index
            int j = KNN_ - 1;
            while (j > 0 && d < bv[j - 1]) { bv[j] = bv[j - 1]; bi[j] = bi[j - 1]; --j; }
            bv[j] = d; bi[j] = c;
        }
    }
#pragma unroll
    for (int j = 0; j < KNN_; ++j) kidx[idx * 8 + j] = bi[j];
    kidx[idx * 8 + 6] = 0; kidx[idx * 8 + 7] = 0;      // pad
}

// =====================================================================
// WMMA GEMM: C(MxNc) = A(MxK)f16 @ B(KxNc)f16 (+bias) [relu]
// B is pre-transposed (N-major: Bt[n*Kd + k]) -> contiguous v16h fragments.
// Each wave: one 16-col N-tile x two 16-row M-tiles (B fragment reused by
// two independent WMMAs). Block = 8 waves = 128 output columns.
// grid = (Nc/128, M/32)
// =====================================================================
template<bool RELU, bool OUT16, bool OUT32>
__global__ __launch_bounds__(256)
void fe_gemm(const half_t* __restrict__ A, const half_t* __restrict__ Bt,
             const float* __restrict__ bias, float* __restrict__ C32,
             half_t* __restrict__ C16, int M, int Nc, int Kd) {
    const int wave = threadIdx.x >> 5;
    const int lane = threadIdx.x & 31;
    const int hs   = lane >> 4;
    const int l15  = lane & 15;
    const int n0   = (blockIdx.x * 8 + wave) * 16;
    const int m0   = blockIdx.y * 32;
    const int ncol = n0 + l15;
    const half_t* arow0 = A + (size_t)(m0 + l15) * Kd;
    const half_t* arow1 = arow0 + (size_t)16 * Kd;
    const half_t* brow  = Bt + (size_t)ncol * Kd;

    v8f acc0 = {}, acc1 = {};
    for (int k0 = 0; k0 < Kd; k0 += 32) {
        v16h b = *(const v16h*)(brow + k0 + hs * 16);
        v16h a0, a1;
        a0 = *(const v16h*)(arow0 + k0 + hs * 16);
        a1 = *(const v16h*)(arow1 + k0 + hs * 16);
        acc0 = __builtin_amdgcn_wmma_f32_16x16x32_f16(false, a0, false, b,
                                                      (short)0, acc0, false, false);
        acc1 = __builtin_amdgcn_wmma_f32_16x16x32_f16(false, a1, false, b,
                                                      (short)0, acc1, false, false);
    }
    const float bv = bias ? bias[ncol] : 0.0f;
#pragma unroll
    for (int r = 0; r < 8; ++r) {
        float x0 = acc0[r] + bv;
        float x1 = acc1[r] + bv;
        if (RELU) { x0 = fmaxf(x0, 0.0f); x1 = fmaxf(x1, 0.0f); }
        const size_t o0 = (size_t)(m0 + hs * 8 + r) * Nc + ncol;
        const size_t o1 = (size_t)(m0 + 16 + hs * 8 + r) * Nc + ncol;
        if (OUT32) { C32[o0] = x0; C32[o1] = x1; }
        if (OUT16) { C16[o0] = (half_t)x0; C16[o1] = (half_t)x1; }
    }
}

// =====================================================================
// fused attention: one workgroup per (n, head); 256 threads (8 waves)
// LDS: Ss f32 128x128 | Ps f16 128x128 (sparse probs) | Qs,Ks f16 128x32 |
//      Vt f16 32x128 | Sc f32 128x8 (6 scores/row) | kL i32 128x8 | mix wts
// =====================================================================
constexpr size_t OFF_PS  = 128 * 128 * 4;                 // 65536
constexpr size_t OFF_QS  = OFF_PS + 128 * 128 * 2;        // 98304
constexpr size_t OFF_KS  = OFF_QS + 128 * 32 * 2;
constexpr size_t OFF_VT  = OFF_KS + 128 * 32 * 2;
constexpr size_t OFF_SC  = OFF_VT + 128 * 32 * 2;         // 122880
constexpr size_t OFF_KI  = OFF_SC + 128 * 8 * 4;          // 126976
constexpr size_t OFF_MW  = OFF_KI + 128 * 8 * 4;          // 131072
constexpr size_t ATTN_SMEM = OFF_MW + 256;                // 131328

__global__ __launch_bounds__(256)
void fe_attn(const half_t* __restrict__ q16, const half_t* __restrict__ k16,
             const half_t* __restrict__ v16, const float* __restrict__ dist,
             const int* __restrict__ kidx,
             const float* __restrict__ mix1_w, const float* __restrict__ mix1_b,
             const float* __restrict__ mix2_w, const float* __restrict__ mix2_b,
             int layer, half_t* __restrict__ att16) {
    extern __shared__ char smem[];
    float*  Ss = (float*)smem;                      // 64 KB scores
    half_t* Ps = (half_t*)(smem + OFF_PS);          // 32 KB probs (sparse)
    half_t* Qs = (half_t*)(smem + OFF_QS);          // 8 KB
    half_t* Ks = (half_t*)(smem + OFF_KS);          // 8 KB
    half_t* Vt = (half_t*)(smem + OFF_VT);          // 8 KB (32 x 128)
    float*  Sc = (float*)(smem + OFF_SC);           // 4 KB
    int*    kL = (int*)(smem + OFF_KI);             // 4 KB
    float*  mw = (float*)(smem + OFF_MW);

    const int n  = blockIdx.x / H_;
    const int hd = blockIdx.x % H_;
    const int tid  = threadIdx.x;
    const int wave = tid >> 5;
    const int lane = tid & 31;
    const int hs   = lane >> 4;
    const int l15  = lane & 15;

    // ---- mix weights: [0..15]=w_dot [16..31]=w_cost [32..47]=b1 [48..63]=w2 [64]=b2
    if (tid < 65) {
        const int base = (layer * H_ + hd) * MS_;
        if (tid < 16)      mw[tid] = mix1_w[((layer * H_ + hd) * 2 + 0) * MS_ + tid];
        else if (tid < 32) mw[tid] = mix1_w[((layer * H_ + hd) * 2 + 1) * MS_ + (tid - 16)];
        else if (tid < 48) mw[tid] = mix1_b[base + (tid - 32)];
        else if (tid < 64) mw[tid] = mix2_w[base + (tid - 48)];
        else               mw[64]  = mix2_b[layer * H_ + hd];
    }
    // ---- kNN indices for this fleet (128 rows x 8)
    for (int t = tid; t < 1024; t += 256) kL[t] = kidx[n * L_ * 8 + t];
    // ---- zero Ps (sparse fill later)
    {
        const v8h z = {};
        for (int c = tid; c < 2048; c += 256) *(v8h*)(Ps + c * 8) = z;
    }
    // ---- stage Q,K (row-major 128x32) and V transposed (32x128), v8h global loads
    for (int c = tid; c < 512; c += 256) {
        const int l = c >> 2, dcol = (c & 3) * 8;
        const size_t g = (size_t)(n * L_ + l) * D_ + hd * DK_ + dcol;
        v8h qv = *(const v8h*)(q16 + g);
        v8h kv = *(const v8h*)(k16 + g);
        v8h vv = *(const v8h*)(v16 + g);
        *(v8h*)(Qs + l * 32 + dcol) = qv;
        *(v8h*)(Ks + l * 32 + dcol) = kv;
#pragma unroll
        for (int j = 0; j < 8; ++j) Vt[(dcol + j) * 128 + l] = vv[j];
    }
    __syncthreads();

    // ---- S = (Q @ K^T) * inv_sqrt_dk : 64 WMMA tiles, single K-step (DK=32)
#pragma unroll 2
    for (int t8 = 0; t8 < 8; ++t8) {
        const int tt = wave + t8 * 8;
        const int rt = tt >> 3, ct = tt & 7;
        const int m    = rt * 16 + l15;
        const int cidx = ct * 16 + l15;          // K-matrix row == score column
        v16h a;
        a = *(const v16h*)(Qs + m * 32 + hs * 16);
        v16h b = *(const v16h*)(Ks + cidx * 32 + hs * 16);   // B(k,col)=K[col][k]
        v8f c = {};
        c = __builtin_amdgcn_wmma_f32_16x16x32_f16(false, a, false, b,
                                                   (short)0, c, false, false);
#pragma unroll
        for (int r = 0; r < 8; ++r)
            Ss[(rt * 16 + hs * 8 + r) * 128 + ct * 16 + l15] = c[r] * INV_SQRT_DK;
    }
    __syncthreads();

    // ---- sparse score-mix MLP: only the 6 kNN columns per row (768 elems)
    for (int e = tid; e < 128 * KNN_; e += 256) {
        const int r = e / KNN_;
        const int j = e - r * KNN_;
        const int c = kL[r * 8 + j];
        const float dot  = Ss[r * 128 + c];
        const float cost = dist[(size_t)(n * L_ + r) * L_ + c];
        float s = mw[64];
#pragma unroll
        for (int m = 0; m < MS_; ++m) {
            float t1 = fmaxf(mw[m] * dot + mw[16 + m] * cost + mw[32 + m], 0.0f);
            s += mw[48 + m] * t1;
        }
        Sc[r * 8 + j] = s;
    }
    __syncthreads();

    // ---- 6-wide softmax per row, scatter probs into zeroed Ps (f16)
    if (tid < 128) {
        const int r = tid;
        float x[KNN_];
        float mx = -3.4e38f;
#pragma unroll
        for (int j = 0; j < KNN_; ++j) { x[j] = Sc[r * 8 + j]; mx = fmaxf(mx, x[j]); }
        float s = 0.0f;
#pragma unroll
        for (int j = 0; j < KNN_; ++j) { x[j] = expf(x[j] - mx); s += x[j]; }
        const float inv = 1.0f / s;
#pragma unroll
        for (int j = 0; j < KNN_; ++j)
            Ps[r * 128 + kL[r * 8 + j]] = (half_t)(x[j] * inv);
    }
    __syncthreads();

    // ---- O = P(128x128) @ V(128x32): 16 tiles x 4 K-steps, Vt contiguous B frags
    for (int tt = wave; tt < 16; tt += 8) {
        const int rt = tt >> 1, ct = tt & 1;
        const int m    = rt * 16 + l15;
        const int colB = ct * 16 + l15;
        v8f acc = {};
#pragma unroll
        for (int k0 = 0; k0 < 128; k0 += 32) {
            v16h a;
            a = *(const v16h*)(Ps + m * 128 + k0 + hs * 16);
            v16h b = *(const v16h*)(Vt + colB * 128 + k0 + hs * 16);
            acc = __builtin_amdgcn_wmma_f32_16x16x32_f16(false, a, false, b,
                                                         (short)0, acc, false, false);
        }
#pragma unroll
        for (int r = 0; r < 8; ++r) {
            const int mrow = rt * 16 + hs * 8 + r;
            att16[(size_t)(n * L_ + mrow) * D_ + hd * DK_ + colB] = (half_t)acc[r];
        }
    }
}

// =====================================================================
// residual + LayerNorm: out = LN(a + b) * g + beta   (one row per block)
// =====================================================================
__global__ __launch_bounds__(256)
void fe_add_ln(const float* __restrict__ a, const float* __restrict__ b,
               const float* __restrict__ g, const float* __restrict__ beta,
               float* __restrict__ out32, half_t* __restrict__ out16) {
    const int row = blockIdx.x;
    const int t   = threadIdx.x;
    const size_t o = (size_t)row * D_ + t;
    const float x = a[o] + b[o];
    __shared__ float red[256];
    red[t] = x;
    __syncthreads();
    for (int s = 128; s > 0; s >>= 1) {
        if (t < s) red[t] += red[t + s];
        __syncthreads();
    }
    const float mean = red[0] / (float)D_;
    __syncthreads();
    const float d = x - mean;
    red[t] = d * d;
    __syncthreads();
    for (int s = 128; s > 0; s >>= 1) {
        if (t < s) red[t] += red[t + s];
        __syncthreads();
    }
    const float var = red[0] / (float)D_;
    const float y = d * rsqrtf(var + 1e-5f) * g[t] + beta[t];
    out32[o] = y;
    out16[o] = (half_t)y;
}

// =====================================================================
// host-side launcher
// =====================================================================
extern "C" void kernel_launch(void* const* d_in, const int* in_sizes, int n_in,
                              void* d_out, int out_size, void* d_ws, size_t ws_size,
                              hipStream_t stream) {
    (void)in_sizes; (void)n_in; (void)out_size; (void)ws_size;

    const float* vehicles = (const float*)d_in[0];
    const float* fe       = (const float*)d_in[1];
    const float* Wproj    = (const float*)d_in[2];
    const float* bproj    = (const float*)d_in[3];
    const float* Wq       = (const float*)d_in[4];
    const float* Wk       = (const float*)d_in[5];
    const float* Wv       = (const float*)d_in[6];
    const float* mix1_w   = (const float*)d_in[7];
    const float* mix1_b   = (const float*)d_in[8];
    const float* mix2_w   = (const float*)d_in[9];
    const float* mix2_b   = (const float*)d_in[10];
    const float* Wo       = (const float*)d_in[11];
    const float* bo       = (const float*)d_in[12];
    const float* n1_g     = (const float*)d_in[13];
    const float* n1_b     = (const float*)d_in[14];
    const float* ff1_w    = (const float*)d_in[15];
    const float* ff1_b    = (const float*)d_in[16];
    const float* ff2_w    = (const float*)d_in[17];
    const float* ff2_b    = (const float*)d_in[18];
    const float* n2_g     = (const float*)d_in[19];
    const float* n2_b     = (const float*)d_in[20];
    float* outp           = (float*)d_out;

    // ---- workspace carve (256-B aligned) ----
    char* ws = (char*)d_ws;
    size_t off = 0;
    auto carve = [&](size_t bytes) -> char* {
        char* p = ws + off;
        off = (off + bytes + 255) & ~(size_t)255;
        return p;
    };
    float*    h32    = (float*)   carve((size_t)R_ * D_ * 4);
    float*    h1_32  = (float*)   carve((size_t)R_ * D_ * 4);
    float*    tmp32  = (float*)   carve((size_t)R_ * D_ * 4);
    float*    dist   = (float*)   carve((size_t)N_ * L_ * L_ * 4);
    half_t*   h16    = (half_t*)  carve((size_t)R_ * D_ * 2);
    half_t*   h1_16  = (half_t*)  carve((size_t)R_ * D_ * 2);
    half_t*   att16  = (half_t*)  carve((size_t)R_ * D_ * 2);
    half_t*   q16    = (half_t*)  carve((size_t)R_ * D_ * 2);
    half_t*   k16    = (half_t*)  carve((size_t)R_ * D_ * 2);
    half_t*   v16    = (half_t*)  carve((size_t)R_ * D_ * 2);
    half_t*   ff16   = (half_t*)  carve((size_t)R_ * FF_ * 2);
    half_t*   wq16   = (half_t*)  carve((size_t)LC_ * D_ * D_ * 2);   // transposed
    half_t*   wk16   = (half_t*)  carve((size_t)LC_ * D_ * D_ * 2);   // transposed
    half_t*   wv16   = (half_t*)  carve((size_t)LC_ * D_ * D_ * 2);   // transposed
    half_t*   wo16   = (half_t*)  carve((size_t)LC_ * D_ * D_ * 2);   // transposed
    half_t*   ff1w16 = (half_t*)  carve((size_t)LC_ * D_ * FF_ * 2);  // transposed (FF x D)
    half_t*   ff2w16 = (half_t*)  carve((size_t)LC_ * FF_ * D_ * 2);  // transposed (D x FF)
    int*      kidx   = (int*)     carve((size_t)N_ * L_ * 8 * 4);

    auto cvt_t = [&](const float* s, half_t* d, int Kd, int Nc) {
        dim3 g((Kd * Nc + 255) / 256, LC_);
        fe_convert_f16_t<<<g, 256, 0, stream>>>(s, d, Kd, Nc);
    };
    cvt_t(Wq, wq16, D_, D_);
    cvt_t(Wk, wk16, D_, D_);
    cvt_t(Wv, wv16, D_, D_);
    cvt_t(Wo, wo16, D_, D_);
    cvt_t(ff1_w, ff1w16, D_, FF_);
    cvt_t(ff2_w, ff2w16, FF_, D_);

    fe_pack_dist<<<(N_ * L_ * L_ + 255) / 256, 256, 0, stream>>>(fe, dist);
    fe_proj<<<(R_ * D_ + 255) / 256, 256, 0, stream>>>(vehicles, Wproj, bproj, h32, h16);
    fe_knn<<<(N_ * L_ + 255) / 256, 256, 0, stream>>>(dist, kidx);

    const dim3 blk(256);
    const dim3 grd_dd(D_ / 128, R_ / 32);     // Nc=256, 2 M-tiles/wave
    const dim3 grd_ff(FF_ / 128, R_ / 32);    // Nc=1024

    for (int i = 0; i < LC_; ++i) {
        // Q, K, V projections (f16 out, no bias)
        fe_gemm<false, true, false><<<grd_dd, blk, 0, stream>>>(
            h16, wq16 + (size_t)i * D_ * D_, nullptr, nullptr, q16, R_, D_, D_);
        fe_gemm<false, true, false><<<grd_dd, blk, 0, stream>>>(
            h16, wk16 + (size_t)i * D_ * D_, nullptr, nullptr, k16, R_, D_, D_);
        fe_gemm<false, true, false><<<grd_dd, blk, 0, stream>>>(
            h16, wv16 + (size_t)i * D_ * D_, nullptr, nullptr, v16, R_, D_, D_);

        // fused attention per (n, head)
        fe_attn<<<N_ * H_, blk, ATTN_SMEM, stream>>>(
            q16, k16, v16, dist, kidx, mix1_w, mix1_b, mix2_w, mix2_b, i, att16);

        // output projection + residual + LN1
        fe_gemm<false, false, true><<<grd_dd, blk, 0, stream>>>(
            att16, wo16 + (size_t)i * D_ * D_, bo + (size_t)i * D_, tmp32, nullptr, R_, D_, D_);
        fe_add_ln<<<R_, blk, 0, stream>>>(h32, tmp32, n1_g + (size_t)i * D_,
                                          n1_b + (size_t)i * D_, h1_32, h1_16);

        // feed-forward + residual + LN2
        fe_gemm<true, true, false><<<grd_ff, blk, 0, stream>>>(
            h1_16, ff1w16 + (size_t)i * D_ * FF_, ff1_b + (size_t)i * FF_,
            nullptr, ff16, R_, FF_, D_);
        fe_gemm<false, false, true><<<grd_dd, blk, 0, stream>>>(
            ff16, ff2w16 + (size_t)i * FF_ * D_, ff2_b + (size_t)i * D_,
            tmp32, nullptr, R_, D_, FF_);
        float* out32 = (i == LC_ - 1) ? outp : h32;
        fe_add_ln<<<R_, blk, 0, stream>>>(h1_32, tmp32, n2_g + (size_t)i * D_,
                                          n2_b + (size_t)i * D_, out32, h16);
        // note: with K=6 every row keeps exactly 6 neighbors, so
        // row_fully_masked is identically false -> final zeroing is a no-op.
    }
}
